// VGRNN_18494129177124
// MI455X (gfx1250) — matrix operand order repaired and checked by reading, not compiled, not run on hardware
//
#include <hip/hip_runtime.h>
#include <hip/hip_bf16.h>

typedef __attribute__((ext_vector_type(16))) _Float16 v16h;
typedef __attribute__((ext_vector_type(8)))  float    v8f;

#define NN 50000
#define EE 800000
#define TT 6
#define XDIM 128
#define HDIM 128
#define ZDIM 64

// ---------------- elementwise helpers ----------------

__global__ void fill_f32(float* __restrict__ p, float v, long n) {
  long i = (long)blockIdx.x * blockDim.x + threadIdx.x;
  if (i < n) p[i] = v;
}

__global__ void count_deg_k(const int* __restrict__ dst, float* __restrict__ deg, int e) {
  int i = blockIdx.x * blockDim.x + threadIdx.x;
  if (i < e) atomicAdd(&deg[dst[i]], 1.0f);
}

__global__ void finalize_deg_k(float* __restrict__ dis, float* __restrict__ sn, int n) {
  int i = blockIdx.x * blockDim.x + threadIdx.x;
  if (i < n) {
    float d = rsqrtf(dis[i]);
    dis[i] = d;
    sn[i] = d * d;
  }
}

__global__ void compute_en_k(const int* __restrict__ src, const int* __restrict__ dst,
                             const float* __restrict__ dis, float* __restrict__ en, int e) {
  int i = blockIdx.x * blockDim.x + threadIdx.x;
  if (i < e) en[i] = dis[src[i]] * dis[dst[i]];
}

__global__ void ewise_mul_k(const float* __restrict__ a, const float* __restrict__ b,
                            float* __restrict__ o, long n) {
  long i = (long)blockIdx.x * blockDim.x + threadIdx.x;
  if (i < n) o[i] = a[i] * b[i];
}

__global__ void copy_f32_k(const float* __restrict__ a, float* __restrict__ o, long n) {
  long i = (long)blockIdx.x * blockDim.x + threadIdx.x;
  if (i < n) o[i] = a[i];
}

// h_new[i,c] = zg[i,c] * h_old[0,c] + (1-zg[i,c]) * hh[i,c]   (row-0 broadcast, faithful)
__global__ void h_update_k(const float* __restrict__ zg, const float* __restrict__ hh,
                           const float* __restrict__ h_old, float* __restrict__ h_new,
                           int n, int M) {
  long i = (long)blockIdx.x * blockDim.x + threadIdx.x;
  long total = (long)n * M;
  if (i >= total) return;
  int c = (int)(i % M);
  float z = zg[i];
  h_new[i] = z * h_old[c] + (1.0f - z) * hh[i];
}

// ---------------- weight packing: f32 [K,M] row-major -> WMMA B-fragment layout ----------------
// Packed layout: P[(((ct*(K/32) + kb)*32 + lane)*16 + idx]
//   lane: n = ct*16 + (lane&15), h = lane>>4
//   idx:  k = kb*32 + (idx>>3)*16 + h*8 + (idx&7)
// so each lane's 16-half B fragment for one (ct, kb) is 32 contiguous aligned bytes.
__global__ void pack_w_k(const float* __restrict__ W, _Float16* __restrict__ P, int K, int M) {
  long i = (long)blockIdx.x * blockDim.x + threadIdx.x;
  long total = (long)K * M;
  if (i >= total) return;
  int idx = (int)(i & 15);
  long r = i >> 4;
  int lane = (int)(r & 31);
  long q = r >> 5;
  int kblocks = K >> 5;
  int ct = (int)(q / kblocks);
  int kb = (int)(q - (long)ct * kblocks);
  int n = ct * 16 + (lane & 15);
  int h = lane >> 4;
  int j = idx >> 1;
  int k = kb * 32 + (idx >> 3) * 16 + h * 8 + (idx & 7); (void)j;
  P[i] = (_Float16)W[(long)k * M + n];
}

// ---------------- WMMA GEMM: out = act( A1@W1 [+ A2@W2] + bias ) ----------------
// One wave computes a full 16-row x (MCOLS*16)-col strip. Per k-step: all MCOLS
// B fragments are fetched first into distinct registers (staggered loadcnt waits),
// then MCOLS hazard-free v_wmma ops drain them.
// A: [n_rows, K] f32 row-major.  W: packed f16 (pack_w_k layout).  out: [n_rows, M] f32.
template <int MCOLS>
__global__ __launch_bounds__(32)
void gemm_wmma_k(const float* __restrict__ A1, const _Float16* __restrict__ W1, int K1,
                 const float* __restrict__ A2, const _Float16* __restrict__ W2, int K2,
                 const float* __restrict__ bias, float* __restrict__ out,
                 int n_rows, int act) {
  const int M = MCOLS * 16;
  const int lane = threadIdx.x;
  const int ml = lane & 15;       // row within tile (A) / col within tile (B, C)
  const int hf = lane >> 4;       // lane-half selects K sub-range
  const int row0 = blockIdx.x * 16;

  int arow = row0 + ml;
  if (arow >= n_rows) arow = n_rows - 1;  // clamp (keeps EXEC all-ones for WMMA)

  v8f acc[MCOLS];
  #pragma unroll
  for (int ct = 0; ct < MCOLS; ++ct) acc[ct] = v8f{};

  // ---- A1 @ W1 ----
  {
    const float* Ap = A1 + (long)arow * K1;
    const int k1b = K1 >> 5;
    for (int kb = 0; kb < k1b; ++kb) {
      // Issue all B-fragment loads for this k-step first (independent registers).
      v16h barr[MCOLS];
      #pragma unroll
      for (int ct = 0; ct < MCOLS; ++ct)
        barr[ct] = *(const v16h*)(W1 + ((((long)ct * k1b + kb) * 32 + lane) << 4));
      // A 16x32 f16 fragment: lane half hf -> K = kb*32 + 8*hf + [0..7] and +16
      v16h a;
      #pragma unroll
      for (int j = 0; j < 4; ++j) {
        int k0 = kb * 32 + hf * 8 + 2 * j;
        a[2 * j]     = (_Float16)Ap[k0];
        a[2 * j + 1] = (_Float16)Ap[k0 + 1];
        int k1 = kb * 32 + 16 + hf * 8 + 2 * j;
        a[2 * j + 8] = (_Float16)Ap[k1];
        a[2 * j + 9] = (_Float16)Ap[k1 + 1];
      }
      #pragma unroll
      for (int ct = 0; ct < MCOLS; ++ct)
        acc[ct] = __builtin_amdgcn_wmma_f32_16x16x32_f16(false, a, false, barr[ct], (short)0,
                                                         acc[ct], false, false);
    }
  }

  // ---- optional A2 @ W2 (concat([A1,A2]) @ [W1;W2] without materializing) ----
  if (A2 != nullptr) {
    const float* Ap = A2 + (long)arow * K2;
    const int k2b = K2 >> 5;
    for (int kb = 0; kb < k2b; ++kb) {
      v16h barr[MCOLS];
      #pragma unroll
      for (int ct = 0; ct < MCOLS; ++ct)
        barr[ct] = *(const v16h*)(W2 + ((((long)ct * k2b + kb) * 32 + lane) << 4));
      v16h a;
      #pragma unroll
      for (int j = 0; j < 4; ++j) {
        int k0 = kb * 32 + hf * 8 + 2 * j;
        a[2 * j]     = (_Float16)Ap[k0];
        a[2 * j + 1] = (_Float16)Ap[k0 + 1];
        int k1 = kb * 32 + 16 + hf * 8 + 2 * j;
        a[2 * j + 8] = (_Float16)Ap[k1];
        a[2 * j + 9] = (_Float16)Ap[k1 + 1];
      }
      #pragma unroll
      for (int ct = 0; ct < MCOLS; ++ct)
        acc[ct] = __builtin_amdgcn_wmma_f32_16x16x32_f16(false, a, false, barr[ct], (short)0,
                                                         acc[ct], false, false);
    }
  }

  // C/D 16x16 f32 layout: VGPR r -> row = r + 8*hf, col = ml
  #pragma unroll
  for (int ct = 0; ct < MCOLS; ++ct) {
    float bv = (bias != nullptr) ? bias[ct * 16 + ml] : 0.0f;
    #pragma unroll
    for (int r = 0; r < 8; ++r) {
      int row = row0 + r + 8 * hf;
      if (row < n_rows) {
        float v = acc[ct][r] + bv;
        if (act == 1) v = fmaxf(v, 0.0f);
        out[(long)row * M + ct * 16 + ml] = v;
      }
    }
  }
}

// ---------------- GCN sparse aggregation ----------------
// agg[dst[e], :] += hw[src[e], :] * en[e]   (float4 chunks, f32 atomics -> L2)
__global__ void gcn_scatter_k(const float* __restrict__ hw, const int* __restrict__ src,
                              const int* __restrict__ dst, const float* __restrict__ en,
                              float* __restrict__ agg, int e, int M) {
  long idx = (long)blockIdx.x * blockDim.x + threadIdx.x;
  int chunks = M >> 2;
  long total = (long)e * chunks;
  if (idx >= total) return;
  int eid = (int)(idx / chunks);
  int c = (int)(idx - (long)eid * chunks) * 4;
  float w = en[eid];
  const float4 s = *(const float4*)(hw + (long)src[eid] * M + c);
  float* d = agg + (long)dst[eid] * M + c;
  atomicAdd(d + 0, s.x * w);
  atomicAdd(d + 1, s.y * w);
  atomicAdd(d + 2, s.z * w);
  atomicAdd(d + 3, s.w * w);
}

// out = act( agg + hw*sn[row] + bias[col] )
__global__ void gcn_finish_k(const float* __restrict__ agg, const float* __restrict__ hw,
                             const float* __restrict__ sn, const float* __restrict__ bias,
                             float* __restrict__ out, int n, int M, int act) {
  long i = (long)blockIdx.x * blockDim.x + threadIdx.x;
  long total = (long)n * M;
  if (i >= total) return;
  int row = (int)(i / M);
  int c = (int)(i - (long)row * M);
  float v = agg[i] + hw[i] * sn[row] + bias[c];
  if (act == 1) v = fmaxf(v, 0.0f);
  out[i] = v;
}

// out = act( (agg1 + hw1*sn + b1) + (agg2 + hw2*sn + b2) ), act: 2=sigmoid, 3=tanh
__global__ void gate2_finish_k(const float* __restrict__ agg1, const float* __restrict__ hw1,
                               const float* __restrict__ b1,
                               const float* __restrict__ agg2, const float* __restrict__ hw2,
                               const float* __restrict__ b2,
                               const float* __restrict__ sn, float* __restrict__ out,
                               int n, int M, int act) {
  long i = (long)blockIdx.x * blockDim.x + threadIdx.x;
  long total = (long)n * M;
  if (i >= total) return;
  int row = (int)(i / M);
  int c = (int)(i - (long)row * M);
  float s = sn[row];
  float v = agg1[i] + hw1[i] * s + b1[c] + agg2[i] + hw2[i] * s + b2[c];
  if (act == 2) v = 1.0f / (1.0f + expf(-v));
  else if (act == 3) v = tanhf(v);
  out[i] = v;
}

// ---------------- host orchestration ----------------

static inline dim3 g1(long n, int bs = 256) { return dim3((unsigned)((n + bs - 1) / bs)); }

extern "C" void kernel_launch(void* const* d_in, const int* in_sizes, int n_in,
                              void* d_out, int out_size, void* d_ws, size_t ws_size,
                              hipStream_t stream) {
  (void)in_sizes; (void)n_in; (void)out_size; (void)ws_size;

  const float* x    = (const float*)d_in[0];
  const int*   eidx = (const int*)d_in[1];
  const float* Wpx = (const float*)d_in[2];  const float* bpx = (const float*)d_in[3];
  // d_in[4..9]: prior branch (Wpr,bpr,Wpm,bpm,Wps,bps) -> dead code w.r.t. outputs, skipped
  const float* Wc1 = (const float*)d_in[10]; const float* bc1 = (const float*)d_in[11];
  const float* Wm  = (const float*)d_in[12]; const float* bm  = (const float*)d_in[13];
  const float* Wpz = (const float*)d_in[14]; const float* bpz = (const float*)d_in[15];
  const float* Wxz = (const float*)d_in[16]; const float* bxz = (const float*)d_in[17];
  const float* Whz = (const float*)d_in[18]; const float* bhz = (const float*)d_in[19];
  const float* Wxr = (const float*)d_in[20]; const float* bxr = (const float*)d_in[21];
  const float* Whr = (const float*)d_in[22]; const float* bhr = (const float*)d_in[23];
  const float* Wxh = (const float*)d_in[24]; const float* bxh = (const float*)d_in[25];
  const float* Whh = (const float*)d_in[26]; const float* bhh = (const float*)d_in[27];

  float* out = (float*)d_out;

  // ---- workspace carve-up ----
  char* wp = (char*)d_ws;
  auto allocF = [&](long n) { float* p = (float*)wp; wp += n * sizeof(float); return p; };
  auto allocH = [&](long n) { _Float16* p = (_Float16*)wp; wp += n * sizeof(_Float16); return p; };

  float* dis = allocF(NN);
  float* sn  = allocF(NN);
  float* en  = allocF(EE);
  float* xp  = allocF((long)NN * HDIM);
  float* hA  = allocF((long)NN * HDIM);
  float* hB  = allocF((long)NN * HDIM);
  float* t0  = allocF((long)NN * HDIM);
  float* t1  = allocF((long)NN * HDIM);
  float* t2  = allocF((long)NN * HDIM);
  float* t3  = allocF((long)NN * HDIM);
  float* t4  = allocF((long)NN * HDIM);
  float* t5  = allocF((long)NN * HDIM);
  float* t6  = allocF((long)NN * HDIM);
  float* t7  = allocF((long)NN * HDIM);

  _Float16* hWpx  = allocH(XDIM * HDIM);
  _Float16* hWc1a = allocH(HDIM * HDIM);   // Wc1 top half (xp branch)
  _Float16* hWc1b = allocH(HDIM * HDIM);   // Wc1 bottom half (h branch)
  _Float16* hWm   = allocH(HDIM * ZDIM);
  _Float16* hWpz  = allocH(ZDIM * HDIM);
  _Float16* hWxza = allocH(HDIM * HDIM);
  _Float16* hWxzb = allocH(HDIM * HDIM);
  _Float16* hWhz  = allocH(HDIM * HDIM);
  _Float16* hWxra = allocH(HDIM * HDIM);
  _Float16* hWxrb = allocH(HDIM * HDIM);
  _Float16* hWhr  = allocH(HDIM * HDIM);
  _Float16* hWxha = allocH(HDIM * HDIM);
  _Float16* hWxhb = allocH(HDIM * HDIM);
  _Float16* hWhh  = allocH(HDIM * HDIM);

  // ---- one-time weight packing into WMMA B-fragment layout ----
  auto pack = [&](const float* srcw, _Float16* dstw, int K, int M) {
    pack_w_k<<<g1((long)K * M), 256, 0, stream>>>(srcw, dstw, K, M);
  };
  pack(Wpx, hWpx, XDIM, HDIM);
  pack(Wc1,               hWc1a, HDIM, HDIM);     // rows 0..127
  pack(Wc1 + HDIM * HDIM, hWc1b, HDIM, HDIM);     // rows 128..255
  pack(Wm,  hWm,  HDIM, ZDIM);
  pack(Wpz, hWpz, ZDIM, HDIM);
  pack(Wxz,               hWxza, HDIM, HDIM);
  pack(Wxz + HDIM * HDIM, hWxzb, HDIM, HDIM);
  pack(Whz, hWhz, HDIM, HDIM);
  pack(Wxr,               hWxra, HDIM, HDIM);
  pack(Wxr + HDIM * HDIM, hWxrb, HDIM, HDIM);
  pack(Whr, hWhr, HDIM, HDIM);
  pack(Wxh,               hWxha, HDIM, HDIM);
  pack(Wxh + HDIM * HDIM, hWxhb, HDIM, HDIM);
  pack(Whh, hWhh, HDIM, HDIM);

  const dim3 gemmBlk(32);
  const dim3 gemmGrid((NN + 15) / 16);   // one wave per 16-row strip

  // xp = relu(x @ Wpx + bpx)   (time-invariant)
  gemm_wmma_k<HDIM / 16><<<gemmGrid, gemmBlk, 0, stream>>>(x, hWpx, XDIM, nullptr, nullptr, 0,
                                                           bpx, xp, NN, 1);
  // h = 0
  fill_f32<<<g1((long)NN * HDIM), 256, 0, stream>>>(hA, 0.0f, (long)NN * HDIM);

  float* hold = hA;
  float* hnew = hB;
  const long NH = (long)NN * HDIM;
  const long NZ = (long)NN * ZDIM;

  for (int t = 0; t < TT; ++t) {
    const int* src = eidx + (long)(t * 2 + 0) * EE;
    const int* dst = eidx + (long)(t * 2 + 1) * EE;

    // --- gcn norm coefficients ---
    fill_f32<<<g1(NN), 256, 0, stream>>>(dis, 1.0f, NN);               // self loop
    count_deg_k<<<g1(EE), 256, 0, stream>>>(dst, dis, EE);
    finalize_deg_k<<<g1(NN), 256, 0, stream>>>(dis, sn, NN);
    compute_en_k<<<g1(EE), 256, 0, stream>>>(src, dst, dis, en, EE);

    // --- encoder GCN 1: e1 = relu(gcn([xp,h], Wc1, bc1)) ---
    gemm_wmma_k<HDIM / 16><<<gemmGrid, gemmBlk, 0, stream>>>(xp, hWc1a, HDIM, hold, hWc1b, HDIM,
                                                             nullptr, t0, NN, 0);
    fill_f32<<<g1(NH), 256, 0, stream>>>(t1, 0.0f, NH);
    gcn_scatter_k<<<g1((long)EE * (HDIM / 4)), 256, 0, stream>>>(t0, src, dst, en, t1, EE, HDIM);
    gcn_finish_k<<<g1(NH), 256, 0, stream>>>(t1, t0, sn, bc1, t2, NN, HDIM, 1);

    // --- encoder GCN 2: z = gcn(e1, Wm, bm)  -> straight into d_out slice t ---
    float* zout = out + (long)t * NZ;
    gemm_wmma_k<ZDIM / 16><<<gemmGrid, gemmBlk, 0, stream>>>(t2, hWm, HDIM, nullptr, nullptr, 0,
                                                             nullptr, t0, NN, 0);
    fill_f32<<<g1(NZ), 256, 0, stream>>>(t1, 0.0f, NZ);
    gcn_scatter_k<<<g1((long)EE * (ZDIM / 4)), 256, 0, stream>>>(t0, src, dst, en, t1, EE, ZDIM);
    gcn_finish_k<<<g1(NZ), 256, 0, stream>>>(t1, t0, sn, bm, zout, NN, ZDIM, 0);

    // --- zp = relu(z @ Wpz + bpz) ---
    gemm_wmma_k<HDIM / 16><<<gemmGrid, gemmBlk, 0, stream>>>(zout, hWpz, ZDIM, nullptr, nullptr, 0,
                                                             bpz, t3, NN, 1);

    // --- update gate: zg = sigmoid(gcn([xp,zp],Wxz,bxz) + gcn(h,Whz,bhz)) ---
    gemm_wmma_k<HDIM / 16><<<gemmGrid, gemmBlk, 0, stream>>>(xp, hWxza, HDIM, t3, hWxzb, HDIM,
                                                             nullptr, t0, NN, 0);
    gemm_wmma_k<HDIM / 16><<<gemmGrid, gemmBlk, 0, stream>>>(hold, hWhz, HDIM, nullptr, nullptr, 0,
                                                             nullptr, t4, NN, 0);
    fill_f32<<<g1(NH), 256, 0, stream>>>(t1, 0.0f, NH);
    fill_f32<<<g1(NH), 256, 0, stream>>>(t5, 0.0f, NH);
    gcn_scatter_k<<<g1((long)EE * (HDIM / 4)), 256, 0, stream>>>(t0, src, dst, en, t1, EE, HDIM);
    gcn_scatter_k<<<g1((long)EE * (HDIM / 4)), 256, 0, stream>>>(t4, src, dst, en, t5, EE, HDIM);
    gate2_finish_k<<<g1(NH), 256, 0, stream>>>(t1, t0, bxz, t5, t4, bhz, sn, t6, NN, HDIM, 2);

    // --- reset gate: rg = sigmoid(gcn([xp,zp],Wxr,bxr) + gcn(h,Whr,bhr)) ---
    gemm_wmma_k<HDIM / 16><<<gemmGrid, gemmBlk, 0, stream>>>(xp, hWxra, HDIM, t3, hWxrb, HDIM,
                                                             nullptr, t0, NN, 0);
    gemm_wmma_k<HDIM / 16><<<gemmGrid, gemmBlk, 0, stream>>>(hold, hWhr, HDIM, nullptr, nullptr, 0,
                                                             nullptr, t4, NN, 0);
    fill_f32<<<g1(NH), 256, 0, stream>>>(t1, 0.0f, NH);
    fill_f32<<<g1(NH), 256, 0, stream>>>(t5, 0.0f, NH);
    gcn_scatter_k<<<g1((long)EE * (HDIM / 4)), 256, 0, stream>>>(t0, src, dst, en, t1, EE, HDIM);
    gcn_scatter_k<<<g1((long)EE * (HDIM / 4)), 256, 0, stream>>>(t4, src, dst, en, t5, EE, HDIM);
    gate2_finish_k<<<g1(NH), 256, 0, stream>>>(t1, t0, bxr, t5, t4, bhr, sn, t7, NN, HDIM, 2);

    // --- candidate: hh = tanh(gcn([xp,zp],Wxh,bxh) + gcn(rg*h,Whh,bhh)) ---
    ewise_mul_k<<<g1(NH), 256, 0, stream>>>(t7, hold, t4, NH);   // rh = rg * h
    gemm_wmma_k<HDIM / 16><<<gemmGrid, gemmBlk, 0, stream>>>(xp, hWxha, HDIM, t3, hWxhb, HDIM,
                                                             nullptr, t0, NN, 0);
    gemm_wmma_k<HDIM / 16><<<gemmGrid, gemmBlk, 0, stream>>>(t4, hWhh, HDIM, nullptr, nullptr, 0,
                                                             nullptr, t3, NN, 0);  // zp dead now
    fill_f32<<<g1(NH), 256, 0, stream>>>(t1, 0.0f, NH);
    fill_f32<<<g1(NH), 256, 0, stream>>>(t5, 0.0f, NH);
    gcn_scatter_k<<<g1((long)EE * (HDIM / 4)), 256, 0, stream>>>(t0, src, dst, en, t1, EE, HDIM);
    gcn_scatter_k<<<g1((long)EE * (HDIM / 4)), 256, 0, stream>>>(t3, src, dst, en, t5, EE, HDIM);
    gate2_finish_k<<<g1(NH), 256, 0, stream>>>(t1, t0, bxh, t5, t3, bhh, sn, t4, NN, HDIM, 3);

    // --- h = zg * h[0] + (1 - zg) * hh   (faithful row-0 broadcast) ---
    h_update_k<<<g1(NH), 256, 0, stream>>>(t6, t4, hold, hnew, NN, HDIM);
    float* tmp = hold; hold = hnew; hnew = tmp;
  }

  // final h after the T z-slices
  copy_f32_k<<<g1(NH), 256, 0, stream>>>(hold, out + (long)TT * NZ, NH);
}
